// MaxedOutSathikNeuralCore_46007689675032
// MI455X (gfx1250) — matrix operand order repaired and checked by reading, not compiled, not run on hardware
//
#include <hip/hip_runtime.h>
#include <hip/hip_bf16.h>
#include <math.h>

// Problem constants (from reference)
constexpr int BB = 8, SS = 2048, DD = 512, EE = 64, KK = 8;
constexpr int TT = BB * SS;     // 16384 tokens
constexpr int GG = 2 * DD;      // 1024 gate hidden
constexpr int HH = 4 * DD;      // 2048 expert hidden

typedef __attribute__((ext_vector_type(16))) __bf16 v16bf;
typedef __attribute__((ext_vector_type(8)))  __bf16 v8bf;
typedef __attribute__((ext_vector_type(8)))  float  v8f;

// ---------------------------------------------------------------------------
// WMMA fragment loaders (V_WMMA_F32_16X16X32_BF16 layouts, CDNA5 ISA 7.12.2)
// A (16x32, MxK): lanes 0-15 -> M=lane, K in [0..7]+[16..23]; lanes 16-31 ->
//                 M=lane-16, K in [8..15]+[24..31].
// B (32x16, KxN): lane -> N=lane&15, K same as A ([0..7]+[16..23] / [8..15]+[24..31]).
// C/D f32: vgpr r, lanes 0-15 -> M=r,N=lane; lanes 16-31 -> M=r+8,N=lane-16.
// ---------------------------------------------------------------------------
__device__ __forceinline__ v16bf load_a_frag(const __bf16* tile, int ld, int lane) {
  const int row = lane & 15;
  const int kb  = (lane & 16) >> 1;          // 0 or 8
  const __bf16* p = tile + (size_t)row * ld + kb;
  v8bf lo = *(const v8bf*)(p);               // K = kb..kb+7      (16B)
  v8bf hi = *(const v8bf*)(p + 16);          // K = kb+16..kb+23  (16B)
  v16bf a;
#pragma unroll
  for (int i = 0; i < 8; ++i) { a[i] = lo[i]; a[i + 8] = hi[i]; }
  return a;
}

__device__ __forceinline__ v16bf load_b_frag(const __bf16* tile, int ld, int lane) {
  const int n  = lane & 15;
  const int kb = (lane & 16) >> 1;
  v16bf b;
#pragma unroll
  for (int i = 0; i < 8; ++i) {
    b[i]     = tile[(size_t)(kb + i) * ld + n];
    b[i + 8] = tile[(size_t)(kb + 16 + i) * ld + n];
  }
  return b;
}

// ---------------------------------------------------------------------------
// f32 -> bf16 conversion (weights/activations staged once per launch)
// ---------------------------------------------------------------------------
__global__ void cvt_bf16_kernel(const float* __restrict__ src,
                                __bf16* __restrict__ dst, size_t n) {
  size_t i = (size_t)blockIdx.x * blockDim.x + threadIdx.x;
  size_t stride = (size_t)gridDim.x * blockDim.x;
  for (; i < n; i += stride) dst[i] = (__bf16)src[i];
}

// ---------------------------------------------------------------------------
// Gate: logits = relu(x @ Wg1 + bg1) @ Wg2 + bg2, fused per 16-token tile.
// 4 waves / block; hidden tile kept in LDS as bf16.
// ---------------------------------------------------------------------------
__global__ __launch_bounds__(128)
void gate_kernel(const __bf16* __restrict__ xb, const __bf16* __restrict__ wg1,
                 const float* __restrict__ bg1, const __bf16* __restrict__ wg2,
                 const float* __restrict__ bg2, float* __restrict__ logits) {
  __shared__ __bf16 xa[16 * DD];    // 16 KB
  __shared__ __bf16 hid[16 * GG];   // 32 KB
  const int t0 = blockIdx.x * 16;
  const int tid = threadIdx.x;
  const int lane = tid & 31, wave = tid >> 5;

  const int CH = DD / 8;            // 64 16-byte chunks per row
  for (int i = tid; i < 16 * CH; i += 128) {
    int r = i / CH, c = i % CH;
    ((v8bf*)xa)[r * CH + c] = ((const v8bf*)(xb + (size_t)(t0 + r) * DD))[c];
  }
  __syncthreads();

  // GEMM1 (16 x 512 x 1024) + bias + ReLU -> hid (bf16)
  for (int nt = wave; nt < GG / 16; nt += 4) {
    const int n0 = nt * 16;
    v8f c = {};
    for (int k0 = 0; k0 < DD; k0 += 32) {
      v16bf a = load_a_frag(xa + k0, DD, lane);
      v16bf b = load_b_frag(wg1 + (size_t)k0 * GG + n0, GG, lane);
      c = __builtin_amdgcn_wmma_f32_16x16x32_bf16(false, a, false, b, (short)0, c, false, false);
    }
    const int col = n0 + (lane & 15);
    const int rbase = (lane & 16) ? 8 : 0;
    const float bias = bg1[col];
#pragma unroll
    for (int r = 0; r < 8; ++r) {
      float v = c[r] + bias;
      hid[(size_t)(rbase + r) * GG + col] = (__bf16)(v > 0.f ? v : 0.f);
    }
  }
  __syncthreads();

  // GEMM2 (16 x 1024 x 64), one 16-col tile per wave
  {
    const int n0 = wave * 16;
    v8f c = {};
    for (int k0 = 0; k0 < GG; k0 += 32) {
      v16bf a = load_a_frag(hid + k0, GG, lane);
      v16bf b = load_b_frag(wg2 + (size_t)k0 * EE + n0, EE, lane);
      c = __builtin_amdgcn_wmma_f32_16x16x32_bf16(false, a, false, b, (short)0, c, false, false);
    }
    const int col = n0 + (lane & 15);
    const int rbase = (lane & 16) ? 8 : 0;
    const float bias = bg2[col];
#pragma unroll
    for (int r = 0; r < 8; ++r)
      logits[(size_t)(t0 + rbase + r) * EE + col] = c[r] + bias;
  }
}

// ---------------------------------------------------------------------------
// Softmax over 64 experts, top-8 select + renormalize, per-expert counts,
// usage accumulation for the load-balancing loss. One thread per token.
// ---------------------------------------------------------------------------
__global__ __launch_bounds__(256)
void route_kernel(const float* __restrict__ logits, int* __restrict__ topk_i,
                  float* __restrict__ topk_w, int* __restrict__ counts,
                  float* __restrict__ usage) {
  __shared__ float us[EE];
  const int tid = threadIdx.x;
  if (tid < EE) us[tid] = 0.f;
  __syncthreads();

  const int t = blockIdx.x * blockDim.x + tid;
  float p[EE];
  float mx = -1e30f;
#pragma unroll
  for (int e = 0; e < EE; ++e) { p[e] = logits[(size_t)t * EE + e]; mx = fmaxf(mx, p[e]); }
  float sum = 0.f;
#pragma unroll
  for (int e = 0; e < EE; ++e) { p[e] = __expf(p[e] - mx); sum += p[e]; }
  const float inv = 1.f / sum;
#pragma unroll
  for (int e = 0; e < EE; ++e) { p[e] *= inv; atomicAdd(&us[e], p[e]); }

  int ids[KK]; float ws[KK]; float wsum = 0.f;
  for (int k = 0; k < KK; ++k) {
    float best = -1.f; int bi = 0;
#pragma unroll
    for (int e = 0; e < EE; ++e) if (p[e] > best) { best = p[e]; bi = e; }
    p[bi] = -1.f;
    ids[k] = bi; ws[k] = best; wsum += best;
  }
  const float winv = 1.f / wsum;
  for (int k = 0; k < KK; ++k) {
    topk_i[(size_t)t * KK + k] = ids[k];
    topk_w[(size_t)t * KK + k] = ws[k] * winv;
    atomicAdd(&counts[ids[k]], 1);
  }
  __syncthreads();
  if (tid < EE) atomicAdd(&usage[tid], us[tid]);
}

// ---------------------------------------------------------------------------
// Serial 64-entry exclusive scan + lb_loss (trivial work).
// ---------------------------------------------------------------------------
__global__ void scan_kernel(const int* __restrict__ counts, int* __restrict__ offsets,
                            const float* __restrict__ usage, float* __restrict__ lb_out) {
  if (threadIdx.x == 0 && blockIdx.x == 0) {
    int acc = 0;
    for (int e = 0; e < EE; ++e) { offsets[e] = acc; acc += counts[e]; }
    float m = 0.f;
    for (int e = 0; e < EE; ++e) {
      float u = usage[e] / (float)TT - 1.0f / (float)EE;
      m += u * u;
    }
    lb_out[0] = 0.01f * (m / (float)EE);
  }
}

// ---------------------------------------------------------------------------
// Scatter (token, k) pairs into per-expert contiguous lists (grouped GEMM).
// ---------------------------------------------------------------------------
__global__ __launch_bounds__(256)
void scatter_kernel(const int* __restrict__ topk_i, const float* __restrict__ topk_w,
                    const int* __restrict__ offsets, int* __restrict__ cursor,
                    int* __restrict__ pair_tok, float* __restrict__ pair_w) {
  const int i = blockIdx.x * blockDim.x + threadIdx.x;
  if (i >= TT * KK) return;
  const int t = i / KK;
  const int e = topk_i[i];
  const int pos = offsets[e] + atomicAdd(&cursor[e], 1);
  pair_tok[pos] = t;
  pair_w[pos] = topk_w[i];
}

// ---------------------------------------------------------------------------
// Fused expert: per block = (32-token tile, expert e).
//   A = gathered x rows (LDS, 32KB bf16)
//   h = gelu(A @ W1e + b1e)    (LDS, 128KB bf16)
//   y = h @ W2e + b2e ; out[token] += y * combine_weight  (f32 atomics)
// 8 waves; all matmuls via v_wmma_f32_16x16x32_bf16.
// ---------------------------------------------------------------------------
__global__ __launch_bounds__(256)
void expert_kernel(const __bf16* __restrict__ xb, const __bf16* __restrict__ w1,
                   const float* __restrict__ b1, const __bf16* __restrict__ w2,
                   const float* __restrict__ b2, const int* __restrict__ counts,
                   const int* __restrict__ offsets, const int* __restrict__ pair_tok,
                   const float* __restrict__ pair_w, float* __restrict__ out) {
  const int e = blockIdx.y;
  const int ne = counts[e];
  const int row0 = blockIdx.x * 32;
  if (row0 >= ne) return;

  extern __shared__ char smem_raw[];
  __bf16* xa  = (__bf16*)smem_raw;        // 32 x 512  bf16 = 32 KB
  __bf16* hb  = xa + 32 * DD;             // 32 x 2048 bf16 = 128 KB
  int*    tok = (int*)(hb + 32 * HH);     // 32 tokens
  float*  wgt = (float*)(tok + 32);       // 32 combine weights

  const int tid = threadIdx.x;
  const int lane = tid & 31, wave = tid >> 5;
  const int base = offsets[e];

  if (tid < 32) {
    const bool valid = (row0 + tid) < ne;
    tok[tid] = valid ? pair_tok[base + row0 + tid] : 0;
    wgt[tid] = valid ? pair_w[base + row0 + tid] : 0.f;
  }
  __syncthreads();

  const int CH = DD / 8;                  // 64 chunks per row
  for (int i = tid; i < 32 * CH; i += 256) {
    int r = i / CH, c = i % CH;
    ((v8bf*)xa)[r * CH + c] = ((const v8bf*)(xb + (size_t)tok[r] * DD))[c];
  }
  __syncthreads();

  const __bf16* w1e = w1 + (size_t)e * DD * HH;
  const float*  b1e = b1 + (size_t)e * HH;

  // GEMM1: [32 x 512] @ [512 x 2048]; wave -> m-tile (wave&3), strided n-tiles
  {
    const int mt = wave & 3;
    for (int nt = (wave >> 2); nt < HH / 16; nt += 2) {
      const int n0 = nt * 16;
      v8f c = {};
      for (int k0 = 0; k0 < DD; k0 += 32) {
        v16bf a = load_a_frag(xa + (size_t)mt * 16 * DD + k0, DD, lane);
        v16bf b = load_b_frag(w1e + (size_t)k0 * HH + n0, HH, lane);
        c = __builtin_amdgcn_wmma_f32_16x16x32_bf16(false, a, false, b, (short)0, c, false, false);
      }
      const int col = n0 + (lane & 15);
      const int rbase = mt * 16 + ((lane & 16) ? 8 : 0);
      const float bias = b1e[col];
#pragma unroll
      for (int r = 0; r < 8; ++r) {
        float v = c[r] + bias;
        v = 0.5f * v * (1.0f + erff(v * 0.70710678118654752f));  // erf-based GELU
        hb[(size_t)(rbase + r) * HH + col] = (__bf16)v;
      }
    }
  }
  __syncthreads();

  const __bf16* w2e = w2 + (size_t)e * HH * DD;
  const float*  b2e = b2 + (size_t)e * DD;

  // GEMM2: [32 x 2048] @ [2048 x 512]; scale by combine weight, scatter-add
  {
    const int mt = wave & 1;
    for (int nt = (wave >> 1); nt < DD / 16; nt += 4) {
      const int n0 = nt * 16;
      v8f c = {};
      for (int k0 = 0; k0 < HH; k0 += 32) {
        v16bf a = load_a_frag(hb + (size_t)mt * 16 * HH + k0, HH, lane);
        v16bf b = load_b_frag(w2e + (size_t)k0 * DD + n0, DD, lane);
        c = __builtin_amdgcn_wmma_f32_16x16x32_bf16(false, a, false, b, (short)0, c, false, false);
      }
      const int col = n0 + (lane & 15);
      const int rb = (lane & 16) ? 8 : 0;
      const float bias = b2e[col];
#pragma unroll
      for (int r = 0; r < 8; ++r) {
        const int lr = mt * 16 + rb + r;
        if (row0 + lr < ne)
          atomicAdd(&out[(size_t)tok[lr] * DD + col], (c[r] + bias) * wgt[lr]);
      }
    }
  }
}

// ---------------------------------------------------------------------------
// Host orchestration
// ---------------------------------------------------------------------------
extern "C" void kernel_launch(void* const* d_in, const int* in_sizes, int n_in,
                              void* d_out, int out_size, void* d_ws, size_t ws_size,
                              hipStream_t stream) {
  const float* x   = (const float*)d_in[0];
  const float* Wg1 = (const float*)d_in[1];
  const float* bg1 = (const float*)d_in[2];
  const float* Wg2 = (const float*)d_in[3];
  const float* bg2 = (const float*)d_in[4];
  const float* W1  = (const float*)d_in[5];
  const float* b1  = (const float*)d_in[6];
  const float* W2  = (const float*)d_in[7];
  const float* b2  = (const float*)d_in[8];
  float* out = (float*)d_out;
  float* lb_out = out + (size_t)TT * DD;

  // Workspace carve-out (~292 MB)
  char* ws = (char*)d_ws;
  auto alloc = [&](size_t bytes) -> char* {
    char* p = ws; ws += (bytes + 255) & ~(size_t)255; return p;
  };
  __bf16* xb   = (__bf16*)alloc((size_t)TT * DD * 2);
  __bf16* wg1b = (__bf16*)alloc((size_t)DD * GG * 2);
  __bf16* wg2b = (__bf16*)alloc((size_t)GG * EE * 2);
  __bf16* w1b  = (__bf16*)alloc((size_t)EE * DD * HH * 2);
  __bf16* w2b  = (__bf16*)alloc((size_t)EE * HH * DD * 2);
  float* logits  = (float*)alloc((size_t)TT * EE * 4);
  int*   topk_i  = (int*)  alloc((size_t)TT * KK * 4);
  float* topk_w  = (float*)alloc((size_t)TT * KK * 4);
  int*   pair_t  = (int*)  alloc((size_t)TT * KK * 4);
  float* pair_w  = (float*)alloc((size_t)TT * KK * 4);
  int*   counts  = (int*)  alloc(EE * 4);
  int*   offsets = (int*)  alloc(EE * 4);
  int*   cursor  = (int*)  alloc(EE * 4);
  float* usage   = (float*)alloc(EE * 4);

  // Re-initialize accumulation state every call (graph-replay safe)
  hipMemsetAsync(d_out, 0, (size_t)out_size * sizeof(float), stream);
  hipMemsetAsync(counts, 0, EE * sizeof(int), stream);
  hipMemsetAsync(cursor, 0, EE * sizeof(int), stream);
  hipMemsetAsync(usage,  0, EE * sizeof(float), stream);

  // Stage bf16 copies (weights read from HBM as f32 exactly once)
  cvt_bf16_kernel<<<2048, 256, 0, stream>>>(x,   xb,   (size_t)TT * DD);
  cvt_bf16_kernel<<<512,  256, 0, stream>>>(Wg1, wg1b, (size_t)DD * GG);
  cvt_bf16_kernel<<<64,   256, 0, stream>>>(Wg2, wg2b, (size_t)GG * EE);
  cvt_bf16_kernel<<<4096, 256, 0, stream>>>(W1,  w1b,  (size_t)EE * DD * HH);
  cvt_bf16_kernel<<<4096, 256, 0, stream>>>(W2,  w2b,  (size_t)EE * HH * DD);

  gate_kernel<<<TT / 16, 128, 0, stream>>>(xb, wg1b, bg1, wg2b, bg2, logits);
  route_kernel<<<TT / 256, 256, 0, stream>>>(logits, topk_i, topk_w, counts, usage);
  scan_kernel<<<1, 1, 0, stream>>>(counts, offsets, usage, lb_out);
  scatter_kernel<<<(TT * KK) / 256, 256, 0, stream>>>(topk_i, topk_w, offsets,
                                                      cursor, pair_t, pair_w);

  const size_t smem = (size_t)32 * DD * 2 + (size_t)32 * HH * 2 + 32 * 4 + 32 * 4;
  expert_kernel<<<dim3(TT / 32, EE), 256, smem, stream>>>(
      xb, w1b, b1, w2b, b2, counts, offsets, pair_t, pair_w, out);
}
